// NGPRadianceField_46763603919057
// MI455X (gfx1250) — matrix operand order repaired and checked by reading, not compiled, not run on hardware
//
#include <hip/hip_runtime.h>
#include <hip/hip_bf16.h>

typedef _Float16 half16 __attribute__((ext_vector_type(16)));
typedef _Float16 half8  __attribute__((ext_vector_type(8)));
typedef float    f32x8  __attribute__((ext_vector_type(8)));

#define HID 256
#define GEO 15
#define POS_LAT 63
#define DIR_LAT 27
#define ACT_S 264   // column stride (halves): 256 rows + 8 pad -> conflict-free b128 stores

// Dynamic LDS layout (bytes):
//   wbuf : [0, 131072)         half[65536]        transposed f16 weights (max 256x256)
//   act  : [131072, 266240)    half[256*ACT_S]    activations, column-major [feat][point]
//   head : [266240, 282624)    float[4096]        256 rows x 16 cols f32 scratch
//   bias : [282624, 285696)    float[768]
static constexpr int SMEM_BYTES = 285696;

// Stage W[K][NSRC] (row-major f32, global) into LDS as wT[n][k] f16 with
// k-padding to PK and n-padding to PN; row stride SK halves.
template<int K, int NSRC, int PK, int PN, int SK>
__device__ __forceinline__ void stage_wT(const float* __restrict__ W,
                                         _Float16* dst, int tid, int nthr) {
  for (int i = tid; i < K * NSRC; i += nthr) {       // coalesced global reads
    int k = i / NSRC, n = i - k * NSRC;
    dst[n * SK + k] = (_Float16)W[i];
  }
  if constexpr (PK > K || PN > NSRC) {
    for (int i = tid; i < PN * PK; i += nthr) {
      int n = i / PK, k = i - n * PK;
      if (k >= K || n >= NSRC) dst[n * SK + k] = (_Float16)0.f;
    }
  }
}

template<int NSRC, int PN>
__device__ __forceinline__ void stage_bias(const float* __restrict__ b,
                                           float* dst, int tid, int nthr) {
  for (int i = tid; i < PN; i += nthr) dst[i] = (i < NSRC) ? b[i] : 0.f;
}

// bias + relu + cvt f16 + single b128 store of one N-tile (C/D rows contiguous
// in the column-major activation buffer).
__device__ __forceinline__ void relu_store(_Float16* act, const f32x8& acc,
                                           float bv, int col, int rowBase) {
  half8 o;
#pragma unroll
  for (int r = 0; r < 8; ++r) {
    float v = acc[r] + bv;
    v = v > 0.f ? v : 0.f;
    o[r] = (_Float16)v;
  }
  *(half8*)(act + col * ACT_S + rowBase) = o;
}

// One dense layer on a wave's private 16-row M-tile.
// A rows live in act (column-major, [k][point], stride ACT_S). B = wT (stride SK).
// MODE 0: out = relu(acc + bias) -> f16, written in place to act (NT*16 cols).
//         Two N-tiles processed concurrently: independent WMMA chains fill each
//         other's WMMA->VALU hazard slots and double matrix-pipe issue density.
// MODE 1: out = acc + bias -> f32 head[row][0..16) (NT must be 1).
template<int KT, int NT, int SK, int MODE>
__device__ __forceinline__ void wmma_layer(_Float16* act, const _Float16* wT,
                                           const float* bias, float* head,
                                           int waveRow, int lane) {
  const int m   = lane & 15;
  const int hi  = lane >> 4;                    // 0 or 1
  const int row = waveRow + m;                  // this lane's A row (point)

  // A k order used here (contiguous, same as B): lanes 0-15 hold K {b..b+15} with b=kt*32,
  // lanes 16-31 K {b+16..b+31}. Column-major act: constant-offset u16 loads.
  half16 A[KT];
#pragma unroll
  for (int kt = 0; kt < KT; ++kt) {
    const _Float16* ap = act + (kt * 32 + hi * 16) * ACT_S + row;
    half16 a;
#pragma unroll
    for (int h = 0; h < 8; ++h) {
      a[h]     = ap[h * ACT_S];
      a[h + 8] = ap[(8 + h) * ACT_S];
    }
    A[kt] = a;
  }

  const int kb = hi * 16;  // B layout: lanes 0-15 K 0..15, lanes 16-31 K 16..31
  const int rowBase = waveRow + hi * 8;

  if constexpr (MODE == 0) {
    static_assert(NT % 2 == 0, "MODE 0 processes N-tiles in pairs");
    for (int nt = 0; nt < NT; nt += 2) {
      const int c0 = (nt + 0) * 16 + m;
      const int c1 = (nt + 1) * 16 + m;
      const _Float16* b0 = wT + c0 * SK + kb;
      const _Float16* b1 = wT + c1 * SK + kb;
      // Double-buffered B for both chains: loads of B[kt+1] issue before the
      // WMMAs consuming B[kt], overlapping LDS latency with matrix work.
      half16 Bc0 = *(const half16*)b0;
      half16 Bc1 = *(const half16*)b1;
      f32x8 acc0 = {0.f, 0.f, 0.f, 0.f, 0.f, 0.f, 0.f, 0.f};
      f32x8 acc1 = {0.f, 0.f, 0.f, 0.f, 0.f, 0.f, 0.f, 0.f};
#pragma unroll
      for (int kt = 0; kt < KT; ++kt) {
        const int ktn = (kt + 1 < KT) ? kt + 1 : kt;
        const half16 Bn0 = *(const half16*)(b0 + ktn * 32);
        const half16 Bn1 = *(const half16*)(b1 + ktn * 32);
        acc0 = __builtin_amdgcn_wmma_f32_16x16x32_f16(false, A[kt], false, Bc0,
                                                      (short)0, acc0, false, false);
        acc1 = __builtin_amdgcn_wmma_f32_16x16x32_f16(false, A[kt], false, Bc1,
                                                      (short)0, acc1, false, false);
        Bc0 = Bn0;
        Bc1 = Bn1;
      }
      relu_store(act, acc0, bias[c0], c0, rowBase);
      relu_store(act, acc1, bias[c1], c1, rowBase);
    }
  } else {
    const _Float16* b0 = wT + m * SK + kb;
    half16 Bc = *(const half16*)b0;
    f32x8 acc = {0.f, 0.f, 0.f, 0.f, 0.f, 0.f, 0.f, 0.f};
#pragma unroll
    for (int kt = 0; kt < KT; ++kt) {
      const int ktn = (kt + 1 < KT) ? kt + 1 : kt;
      const half16 Bn = *(const half16*)(b0 + ktn * 32);
      acc = __builtin_amdgcn_wmma_f32_16x16x32_f16(false, A[kt], false, Bc,
                                                   (short)0, acc, false, false);
      Bc = Bn;
    }
    const float bv = bias[m];
#pragma unroll
    for (int r = 0; r < 8; ++r)
      head[(rowBase + r) * 16 + m] = acc[r] + bv;
  }
}

__global__ void ngp_radiance_kernel(
    const float* __restrict__ pos, const float* __restrict__ dir,
    const float* __restrict__ W1, const float* __restrict__ b1,
    const float* __restrict__ W2, const float* __restrict__ b2,
    const float* __restrict__ W3, const float* __restrict__ b3,
    const float* __restrict__ W4, const float* __restrict__ b4,
    const float* __restrict__ Wc1, const float* __restrict__ bc1,
    const float* __restrict__ Wc2, const float* __restrict__ bc2,
    float* __restrict__ out, int N) {
  extern __shared__ char smem[];
  _Float16* wbuf = (_Float16*)smem;
  _Float16* act  = (_Float16*)(smem + 131072);
  float*    head = (float*)(smem + 266240);
  float*    bias = (float*)(smem + 282624);

  const int tid  = threadIdx.x;
  const int nthr = blockDim.x;        // 512 = 16 wave32
  const int lane = tid & 31;
  const int waveRow = (tid >> 5) * 16;
  const int base = blockIdx.x * 256;

  // ---- stage W1^T (pad K 63->64) + b1; build positional encoding columns ----
  stage_wT<POS_LAT, HID, 64, HID, 64>(W1, wbuf, tid, nthr);
  stage_bias<HID, HID>(b1, bias, tid, nthr);

  if (tid < 256) {
    const int p = base + tid;
    _Float16* acol = act + tid;                // column-major: [feat*ACT_S + point]
    float x[3];
#pragma unroll
    for (int c = 0; c < 3; ++c) {
      x[c] = pos[p * 3 + c] * (1.0f / 1.5f);   // map [-1.5,1.5] -> [-1,1]
      acol[c * ACT_S] = (_Float16)x[c];
    }
#pragma unroll
    for (int s = 0; s < 10; ++s) {
      const float sc = (float)(1 << s);
#pragma unroll
      for (int c = 0; c < 3; ++c) {
        const float a = x[c] * sc;
        acol[(3 + s * 3 + c) * ACT_S]  = (_Float16)__sinf(a);
        acol[(33 + s * 3 + c) * ACT_S] = (_Float16)__cosf(a);
      }
    }
    acol[63 * ACT_S] = (_Float16)0.f;
  }
  __syncthreads();

  // warm L2/L0 with W2 while layer 1 computes (global_prefetch_b8)
  for (int i = tid * 16; i < HID * HID; i += nthr * 16)
    __builtin_prefetch(W2 + i, 0, 1);

  // ---- layer 1: 64(63) -> 256, relu ----
  wmma_layer<2, 16, 64, 0>(act, wbuf, bias, nullptr, waveRow, lane);
  __syncthreads();

  // ---- layer 2: 256 -> 256, relu ----
  stage_wT<HID, HID, HID, HID, HID>(W2, wbuf, tid, nthr);
  stage_bias<HID, HID>(b2, bias, tid, nthr);
  __syncthreads();
  wmma_layer<8, 16, HID, 0>(act, wbuf, bias, nullptr, waveRow, lane);
  __syncthreads();

  // ---- layer 3: 256 -> 256, relu ----
  stage_wT<HID, HID, HID, HID, HID>(W3, wbuf, tid, nthr);
  stage_bias<HID, HID>(b3, bias, tid, nthr);
  __syncthreads();
  wmma_layer<8, 16, HID, 0>(act, wbuf, bias, nullptr, waveRow, lane);
  __syncthreads();

  // ---- stage head weights: W4^T @0, Wc1^T @4096, Wc2^T @12288 ----
  stage_wT<HID, 1 + GEO, HID, 16, HID>(W4, wbuf, tid, nthr);
  stage_wT<GEO + DIR_LAT, 128, 64, 128, 64>(Wc1, wbuf + 4096, tid, nthr);
  stage_wT<128, 3, 128, 16, 128>(Wc2, wbuf + 12288, tid, nthr);
  stage_bias<1 + GEO, 16>(b4, bias, tid, nthr);
  stage_bias<128, 128>(bc1, bias + 256, tid, nthr);
  stage_bias<3, 16>(bc2, bias + 512, tid, nthr);
  __syncthreads();

  // ---- layer 4: 256 -> 16 (density + feat), f32 to head ----
  wmma_layer<8, 1, HID, 1>(act, wbuf, bias, head, waveRow, lane);
  __syncthreads();

  // ---- density out; build color-net input cols [feat(15) | dir-enc(27) | 0] ----
  if (tid < 256) {
    const int p = base + tid;
    const float* hrow = head + tid * 16;
    out[3 * N + p] = __expf(hrow[0] - 1.0f);          // trunc_exp(h0 - 1)
    _Float16* acol = act + tid;
#pragma unroll
    for (int j = 0; j < GEO; ++j) acol[j * ACT_S] = (_Float16)hrow[1 + j];
    float d[3]; float nrm = 0.f;
#pragma unroll
    for (int c = 0; c < 3; ++c) { d[c] = dir[p * 3 + c]; nrm += d[c] * d[c]; }
    const float rn = 1.0f / fmaxf(sqrtf(nrm), 1e-12f);
#pragma unroll
    for (int c = 0; c < 3; ++c) { d[c] *= rn; acol[(GEO + c) * ACT_S] = (_Float16)d[c]; }
#pragma unroll
    for (int s = 0; s < 4; ++s) {
      const float sc = (float)(1 << s);
#pragma unroll
      for (int c = 0; c < 3; ++c) {
        const float a = d[c] * sc;
        acol[(GEO + 3 + s * 3 + c) * ACT_S]  = (_Float16)__sinf(a);
        acol[(GEO + 15 + s * 3 + c) * ACT_S] = (_Float16)__cosf(a);
      }
    }
#pragma unroll
    for (int j = GEO + DIR_LAT; j < 64; ++j) acol[j * ACT_S] = (_Float16)0.f;
  }
  __syncthreads();

  // ---- color layer 1: 64(42) -> 128, relu ----
  wmma_layer<2, 8, 64, 0>(act, wbuf + 4096, bias + 256, nullptr, waveRow, lane);
  __syncthreads();

  // ---- color layer 2: 128 -> 16(3), f32 to head ----
  wmma_layer<4, 1, 128, 1>(act, wbuf + 12288, bias + 512, head, waveRow, lane);
  __syncthreads();

  // ---- rgb = sigmoid ----
  if (tid < 256) {
    const int p = base + tid;
    const float* hrow = head + tid * 16;
#pragma unroll
    for (int c = 0; c < 3; ++c)
      out[3 * p + c] = 1.0f / (1.0f + __expf(-hrow[c]));
  }
}

extern "C" void kernel_launch(void* const* d_in, const int* in_sizes, int n_in,
                              void* d_out, int out_size, void* d_ws, size_t ws_size,
                              hipStream_t stream) {
  const float* pos = (const float*)d_in[0];
  const float* dir = (const float*)d_in[1];
  const float* W1  = (const float*)d_in[2];
  const float* b1  = (const float*)d_in[3];
  const float* W2  = (const float*)d_in[4];
  const float* b2  = (const float*)d_in[5];
  const float* W3  = (const float*)d_in[6];
  const float* b3  = (const float*)d_in[7];
  const float* W4  = (const float*)d_in[8];
  const float* b4  = (const float*)d_in[9];
  const float* Wc1 = (const float*)d_in[10];
  const float* bc1 = (const float*)d_in[11];
  const float* Wc2 = (const float*)d_in[12];
  const float* bc2 = (const float*)d_in[13];
  float* out = (float*)d_out;

  const int N = in_sizes[0] / 3;

  (void)hipFuncSetAttribute((const void*)ngp_radiance_kernel,
                            hipFuncAttributeMaxDynamicSharedMemorySize,
                            SMEM_BYTES);

  dim3 grid(N / 256), block(512);
  ngp_radiance_kernel<<<grid, block, SMEM_BYTES, stream>>>(
      pos, dir, W1, b1, W2, b2, W3, b3, W4, b4, Wc1, bc1, Wc2, bc2, out, N);
}
